// SwinTransformerBlock_39779987095800
// MI455X (gfx1250) — matrix operand order repaired and checked by reading, not compiled, not run on hardware
//
#include <hip/hip_runtime.h>
#include <hip/hip_bf16.h>
#include <math.h>

typedef __attribute__((ext_vector_type(16))) _Float16 v16h;
typedef __attribute__((ext_vector_type(8)))  _Float16 v8h;
typedef __attribute__((ext_vector_type(8)))  float    v8f;

#define NTOK   100352        // B*H*W tokens
#define DIMC   384
#define NHH    12
#define HDIM   32
#define NWTOK  49            // tokens per 7x7 window
#define QE     38535168ull   // 2048*12*49*32 elements per q/k/v plane

static __device__ __forceinline__ v16h cat8(v8h lo, v8h hi) {
  return __builtin_shufflevector(lo, hi, 0,1,2,3,4,5,6,7,8,9,10,11,12,13,14,15);
}
static __device__ __forceinline__ v8f wmma16x16x32(v16h a, v16h b, v8f c) {
  // D = A(16x32 f16) * B(32x16 f16) + C(16x16 f32)
  return __builtin_amdgcn_wmma_f32_16x16x32_f16(false, a, false, b, (short)0, c,
                                                false, false);
}

// ---------------------------------------------------------------------------
// f32 -> f16 conversion (weights)
// ---------------------------------------------------------------------------
__global__ void cvt_f32_f16(const float* __restrict__ in, _Float16* __restrict__ out, int n) {
  int i = blockIdx.x * 256 + threadIdx.x;
  if (i < n) out[i] = (_Float16)in[i];
}

// ---------------------------------------------------------------------------
// LayerNorm over rows of 384; mode 0: row r of `in` (f32, row-major [NTOK,384])
// mode 1: row r is a window-layout token -> gather shifted pixel from x[B,56,56,384]
// ---------------------------------------------------------------------------
__global__ __launch_bounds__(128)
void ln_kernel(const float* __restrict__ in, _Float16* __restrict__ out,
               const float* __restrict__ g, const float* __restrict__ b, int mode) {
  __shared__ float red[128];
  const int r = blockIdx.x, tid = threadIdx.x;
  const float* src;
  if (mode == 0) {
    src = in + (size_t)r * DIMC;
  } else {
    int b_ = r / NWTOK, n = r - b_ * NWTOK;
    int bb = b_ >> 6, wi = b_ & 63;
    int wh = wi >> 3, ww = wi & 7;
    int i = n / 7, j = n - i * 7;
    int ph = (wh * 7 + i + 3) % 56;           // cyclic shift by -3 folded in
    int pw = (ww * 7 + j + 3) % 56;
    src = in + ((((size_t)bb * 56 + ph) * 56) + pw) * DIMC;
  }
  float vals[3];
  float s = 0.f;
#pragma unroll
  for (int t = 0; t < 3; ++t) { vals[t] = src[tid + 128 * t]; s += vals[t]; }
  red[tid] = s; __syncthreads();
  for (int off = 64; off; off >>= 1) { if (tid < off) red[tid] += red[tid + off]; __syncthreads(); }
  float mean = red[0] * (1.0f / DIMC);
  __syncthreads();
  float s2 = 0.f;
#pragma unroll
  for (int t = 0; t < 3; ++t) { float d = vals[t] - mean; s2 += d * d; }
  red[tid] = s2; __syncthreads();
  for (int off = 64; off; off >>= 1) { if (tid < off) red[tid] += red[tid + off]; __syncthreads(); }
  float inv = rsqrtf(red[0] * (1.0f / DIMC) + 1e-5f);
#pragma unroll
  for (int t = 0; t < 3; ++t) {
    int c = tid + 128 * t;
    out[(size_t)r * DIMC + c] = (_Float16)((vals[t] - mean) * inv * g[c] + b[c]);
  }
}

// ---------------------------------------------------------------------------
// Epilogues for the tiled WMMA GEMM
//   0: QKV  -> scale q by 1/sqrt(32), scatter f16 into [which][b_*NH+head][n][d]
//   1: proj -> window-reverse + un-shift + residual(x) -> y (f32)
//   2: fc1  -> exact GELU -> h (f16, row-major N=1536)
//   3: fc2  -> + residual(y) -> d_out (f32)
// ---------------------------------------------------------------------------
template <int EPI>
static __device__ __forceinline__ void epi_store(
    int row, int col, float acc, int N,
    const float* __restrict__ bias, _Float16* __restrict__ out_h,
    float* __restrict__ out_f, const float* __restrict__ resid) {
  float v = acc + bias[col];
  if constexpr (EPI == 0) {
    int which = col / DIMC;
    int cc = col - which * DIMC;
    int head = cc >> 5, d = cc & 31;
    if (which == 0) v *= 0.17677669529663687f;   // 32^-0.5
    int b_ = row / NWTOK, n = row - b_ * NWTOK;
    size_t dst = (size_t)which * QE + (((size_t)b_ * NHH + head) * NWTOK + n) * HDIM + d;
    out_h[dst] = (_Float16)v;
  } else if constexpr (EPI == 1) {
    int b_ = row / NWTOK, n = row - b_ * NWTOK;
    int b = b_ >> 6, wi = b_ & 63;
    int wh = wi >> 3, ww = wi & 7;
    int i = n / 7, j = n - i * 7;
    int ph = (wh * 7 + i + 3) % 56;              // reverse shift (+3)
    int pw = (ww * 7 + j + 3) % 56;
    size_t dst = ((((size_t)b * 56 + ph) * 56) + pw) * DIMC + col;
    out_f[dst] = resid[dst] + v;
  } else if constexpr (EPI == 2) {
    float ge = 0.5f * v * (1.0f + erff(v * 0.70710678118654752f));
    out_h[(size_t)row * N + col] = (_Float16)ge;
  } else {
    size_t dst = (size_t)row * DIMC + col;
    out_f[dst] = resid[dst] + v;
  }
}

// ---------------------------------------------------------------------------
// Tiled WMMA GEMM: C[M,N] = A[M,K] * B[K,N]  (f16 in, f32 acc)
// Block tile 128x64, 8 waves in a 4x2 grid, each wave owns a 32x32 region
// (2x2 WMMA tiles -> 4 v_wmma per BK step). B tile staged TRANSPOSED in LDS
// so every B fragment is two contiguous 16B ds_load_b128 reads.
// M % 128 == 0, N % 64 == 0, K % 32 == 0 (true for all stages here).
// ---------------------------------------------------------------------------
template <int EPI>
__global__ __launch_bounds__(256)
void gemm_wmma_kernel(const _Float16* __restrict__ A, const _Float16* __restrict__ Bw,
                      int M, int N, int K,
                      const float* __restrict__ bias, _Float16* __restrict__ out_h,
                      float* __restrict__ out_f, const float* __restrict__ resid) {
  __shared__ __align__(16) _Float16 As[128][32];   // row-major A tile
  __shared__ __align__(16) _Float16 Bt[64][32];    // TRANSPOSED B tile: Bt[n][k]
  const int tid  = threadIdx.x;
  const int lane = tid & 31, wave = tid >> 5;
  const int m0 = blockIdx.x * 128, n0 = blockIdx.y * 64;
  const int wrow = wave & 3, wcol = wave >> 2;     // 4x2 wave grid (32x32 each)
  const int mrow = lane & 15, hh = lane >> 4;

  // cooperative-load indices
  const int aIdx = tid * 8, ar = aIdx >> 5, ac = aIdx & 31;   // A: 2 planes of 2048 f16
  const int br = aIdx >> 6, bc = aIdx & 63;                   // B: 32x64 tile

  v8f acc00 = {}, acc01 = {}, acc10 = {}, acc11 = {};
  for (int kk = 0; kk < K; kk += 32) {
    __syncthreads();
    // ---- stage A (128x32), vectorized ----
    *(uint4*)&As[ar][ac]      = *(const uint4*)(A + (size_t)(m0 + ar) * K + kk + ac);
    *(uint4*)&As[64 + ar][ac] = *(const uint4*)(A + (size_t)(m0 + 64 + ar) * K + kk + ac);
    // ---- stage B (32x64) transposed into Bt[n][k] ----
    {
      union { uint4 u; _Float16 h[8]; } bu;
      bu.u = *(const uint4*)(Bw + (size_t)(kk + br) * N + n0 + bc);
#pragma unroll
      for (int e = 0; e < 8; ++e) Bt[bc + e][br] = bu.h[e];
    }
    // prefetch next K tile (global_prefetch_b8 path)
    if (kk + 32 < K) {
      __builtin_prefetch(A + (size_t)(m0 + ar) * K + kk + 32 + ac, 0, 3);
      __builtin_prefetch(Bw + (size_t)(kk + 32 + br) * N + n0 + bc, 0, 3);
    }
    __syncthreads();

    // A fragments (two row tiles): contiguous 8-half segments
    const _Float16* ap0 = &As[wrow * 32 + mrow][0];
    const _Float16* ap1 = &As[wrow * 32 + 16 + mrow][0];
    v16h a0 = cat8(*(const v8h*)(ap0 + hh * 16), *(const v8h*)(ap0 + hh * 16 + 8));
    v16h a1 = cat8(*(const v8h*)(ap1 + hh * 16), *(const v8h*)(ap1 + hh * 16 + 8));
    // B fragments (two col tiles) from transposed tile: contiguous reads
    const _Float16* bp0 = &Bt[wcol * 32 + mrow][hh * 16];
    const _Float16* bp1 = &Bt[wcol * 32 + 16 + mrow][hh * 16];
    v16h b0 = cat8(*(const v8h*)bp0, *(const v8h*)(bp0 + 8));
    v16h b1 = cat8(*(const v8h*)bp1, *(const v8h*)(bp1 + 8));

    acc00 = wmma16x16x32(a0, b0, acc00);
    acc01 = wmma16x16x32(a0, b1, acc01);
    acc10 = wmma16x16x32(a1, b0, acc10);
    acc11 = wmma16x16x32(a1, b1, acc11);
  }

#pragma unroll
  for (int r = 0; r < 8; ++r) {
    int gm0 = m0 + wrow * 32 + r + 8 * hh;
    int gm1 = gm0 + 16;
    int gn0 = n0 + wcol * 32 + mrow;
    int gn1 = gn0 + 16;
    epi_store<EPI>(gm0, gn0, acc00[r], N, bias, out_h, out_f, resid);
    epi_store<EPI>(gm0, gn1, acc01[r], N, bias, out_h, out_f, resid);
    epi_store<EPI>(gm1, gn0, acc10[r], N, bias, out_h, out_f, resid);
    epi_store<EPI>(gm1, gn1, acc11[r], N, bias, out_h, out_f, resid);
  }
}

// ---------------------------------------------------------------------------
// Windowed attention, one block per (window, head). N=49 padded to 64, d=32.
// S = Q*K^T (WMMA) + rel-pos-bias + shift mask -> softmax -> O = P*V (WMMA).
// V staged transposed (Vt[d][token]) so P*V B-fragments are contiguous reads.
// ---------------------------------------------------------------------------
__global__ __launch_bounds__(128)
void attn_kernel(const _Float16* __restrict__ q, const _Float16* __restrict__ k,
                 const _Float16* __restrict__ v, const float* __restrict__ rpb,
                 const float* __restrict__ mask, _Float16* __restrict__ out) {
  __shared__ __align__(16) _Float16 Qs[64][32];
  __shared__ __align__(16) _Float16 Ks[64][32];
  __shared__ __align__(16) _Float16 Vt[32][64];    // transposed V: Vt[d][token]
  __shared__ float Sf[64][65];
  __shared__ __align__(16) _Float16 Ps[64][72];

  const int bh = blockIdx.x;
  const int b_ = bh / NHH, head = bh - b_ * NHH;
  const int tid = threadIdx.x, lane = tid & 31, wave = tid >> 5;
  const int mrow = lane & 15, hh = lane >> 4;
  const size_t base = (size_t)bh * NWTOK * HDIM;

  for (int idx = tid; idx < 64 * 32; idx += 128) {
    int r = idx >> 5, c = idx & 31;
    _Float16 zq = (_Float16)0.f, zk = (_Float16)0.f, zv = (_Float16)0.f;
    if (r < NWTOK) {
      zq = q[base + r * HDIM + c];
      zk = k[base + r * HDIM + c];
      zv = v[base + r * HDIM + c];
    }
    Qs[r][c] = zq; Ks[r][c] = zk; Vt[c][r] = zv;
  }
  __syncthreads();

  // ---- S = Q K^T : wave handles row tile tm = wave (K dim = 32, one step) ----
  {
    const _Float16* ap = &Qs[wave * 16 + mrow][0];
    v16h af = cat8(*(const v8h*)(ap + hh * 16), *(const v8h*)(ap + hh * 16 + 8));
#pragma unroll
    for (int tn = 0; tn < 4; ++tn) {
      // B[d][n] = K[n][d]  -> contiguous row slice of Ks (free transpose)
      const _Float16* bp = &Ks[tn * 16 + mrow][hh * 16];
      v16h bf = cat8(*(const v8h*)bp, *(const v8h*)(bp + 8));
      v8f z = {};
      v8f d = wmma16x16x32(af, bf, z);
#pragma unroll
      for (int r = 0; r < 8; ++r)
        Sf[wave * 16 + r + 8 * hh][tn * 16 + mrow] = d[r];
    }
  }
  __syncthreads();

  // ---- softmax rows (bias + mask fused); 3 passes over LDS ----
  if (tid < 64) {
    int m = tid;
    if (m < NWTOK) {
      int my = m / 7, mx = m - my * 7;
      const float* mrowp = mask + (((size_t)(b_ & 63) * NWTOK) + m) * NWTOK;
      float mx_ = -1e30f;
      for (int n = 0; n < NWTOK; ++n) {
        int ny = n / 7, nx = n - ny * 7;
        int ridx = (my - ny + 6) * 13 + (mx - nx + 6);  // RPI(m,n)
        float s = Sf[m][n] + rpb[ridx * NHH + head] + mrowp[n];
        Sf[m][n] = s;
        mx_ = fmaxf(mx_, s);
      }
      float sum = 0.f;
      for (int n = 0; n < NWTOK; ++n) {
        float e = __expf(Sf[m][n] - mx_);
        Sf[m][n] = e; sum += e;
      }
      float inv = 1.0f / sum;
      for (int n = 0; n < NWTOK; ++n) Ps[m][n] = (_Float16)(Sf[m][n] * inv);
      for (int n = NWTOK; n < 64; ++n) Ps[m][n] = (_Float16)0.f;
    } else {
      for (int n = 0; n < 64; ++n) Ps[m][n] = (_Float16)0.f;
    }
  }
  __syncthreads();

  // ---- O = P V : rows tile tm = wave, N=32 (2 tiles), K=64 (2 steps) ----
  {
    v8f acc[2] = {{}, {}};
#pragma unroll
    for (int ks = 0; ks < 2; ++ks) {
      const _Float16* ap = &Ps[wave * 16 + mrow][ks * 32];
      v16h af = cat8(*(const v8h*)(ap + hh * 16), *(const v8h*)(ap + hh * 16 + 8));
#pragma unroll
      for (int tn = 0; tn < 2; ++tn) {
        // B[k][n] = V[token k][d n] = Vt[tn*16+mrow][...] -> contiguous
        const _Float16* bp = &Vt[tn * 16 + mrow][ks * 32 + hh * 16];
        v16h bf = cat8(*(const v8h*)bp, *(const v8h*)(bp + 8));
        acc[tn] = wmma16x16x32(af, bf, acc[tn]);
      }
    }
#pragma unroll
    for (int tn = 0; tn < 2; ++tn) {
#pragma unroll
      for (int r = 0; r < 8; ++r) {
        int m2 = wave * 16 + r + 8 * hh;
        if (m2 < NWTOK)
          out[((size_t)b_ * NWTOK + m2) * DIMC + head * HDIM + tn * 16 + mrow] =
              (_Float16)acc[tn][r];
      }
    }
  }
}

// ---------------------------------------------------------------------------
extern "C" void kernel_launch(void* const* d_in, const int* in_sizes, int n_in,
                              void* d_out, int out_size, void* d_ws, size_t ws_size,
                              hipStream_t stream) {
  const float* x      = (const float*)d_in[0];
  const float* amask  = (const float*)d_in[1];
  const float* g1     = (const float*)d_in[2];
  const float* b1     = (const float*)d_in[3];
  const float* w_qkv  = (const float*)d_in[4];
  const float* b_qkv  = (const float*)d_in[5];
  const float* rpb    = (const float*)d_in[6];
  const float* w_proj = (const float*)d_in[7];
  const float* b_proj = (const float*)d_in[8];
  const float* g2     = (const float*)d_in[9];
  const float* b2     = (const float*)d_in[10];
  const float* w_fc1  = (const float*)d_in[11];
  const float* b_fc1  = (const float*)d_in[12];
  const float* w_fc2  = (const float*)d_in[13];
  const float* b_fc2  = (const float*)d_in[14];
  float* out = (float*)d_out;

  char* ws = (char*)d_ws;
  size_t off = 0;
  auto alloc = [&](size_t bytes) {
    size_t o = off;
    off = (off + bytes + 255) & ~(size_t)255;
    return o;
  };
  const size_t NQKV = (size_t)DIMC * 3 * DIMC;      // 442368
  const size_t NPRJ = (size_t)DIMC * DIMC;          // 147456
  const size_t NF1  = (size_t)DIMC * 4 * DIMC;      // 589824
  const size_t NF2  = NF1;

  _Float16* wqkv16 = (_Float16*)(ws + alloc(NQKV * 2));
  _Float16* wprj16 = (_Float16*)(ws + alloc(NPRJ * 2));
  _Float16* wf1_16 = (_Float16*)(ws + alloc(NF1 * 2));
  _Float16* wf2_16 = (_Float16*)(ws + alloc(NF2 * 2));
  _Float16* xw16   = (_Float16*)(ws + alloc((size_t)NTOK * DIMC * 2));
  _Float16* qkv16  = (_Float16*)(ws + alloc(3ull * QE * 2));
  _Float16* ao16   = (_Float16*)(ws + alloc((size_t)NTOK * DIMC * 2));
  float*    yres   = (float*)   (ws + alloc((size_t)NTOK * DIMC * 4));
  _Float16* yln16  = (_Float16*)(ws + alloc((size_t)NTOK * DIMC * 2));
  _Float16* h16    = (_Float16*)(ws + alloc((size_t)NTOK * 4 * DIMC * 2));

  // 1) weights -> f16
  cvt_f32_f16<<<(int)((NQKV + 255) / 256), 256, 0, stream>>>(w_qkv,  wqkv16, (int)NQKV);
  cvt_f32_f16<<<(int)((NPRJ + 255) / 256), 256, 0, stream>>>(w_proj, wprj16, (int)NPRJ);
  cvt_f32_f16<<<(int)((NF1  + 255) / 256), 256, 0, stream>>>(w_fc1,  wf1_16, (int)NF1);
  cvt_f32_f16<<<(int)((NF2  + 255) / 256), 256, 0, stream>>>(w_fc2,  wf2_16, (int)NF2);

  // 2) LN1 + shift + window partition -> xw16 [NTOK,384] f16
  ln_kernel<<<NTOK, 128, 0, stream>>>(x, xw16, g1, b1, 1);

  // 3) QKV GEMM [NTOK,384]x[384,1152] -> q/k/v planes (q pre-scaled)
  gemm_wmma_kernel<0><<<dim3(NTOK / 128, 1152 / 64), 256, 0, stream>>>(
      xw16, wqkv16, NTOK, 1152, 384, b_qkv, qkv16, nullptr, nullptr);

  // 4) windowed attention (2048 windows x 12 heads)
  attn_kernel<<<2048 * NHH, 128, 0, stream>>>(qkv16, qkv16 + QE, qkv16 + 2 * QE,
                                              rpb, amask, ao16);

  // 5) proj GEMM + window reverse + un-shift + residual -> yres (f32)
  gemm_wmma_kernel<1><<<dim3(NTOK / 128, DIMC / 64), 256, 0, stream>>>(
      ao16, wprj16, NTOK, DIMC, DIMC, b_proj, nullptr, yres, x);

  // 6) LN2 -> yln16
  ln_kernel<<<NTOK, 128, 0, stream>>>(yres, yln16, g2, b2, 0);

  // 7) FC1 + exact GELU -> h16
  gemm_wmma_kernel<2><<<dim3(NTOK / 128, 1536 / 64), 256, 0, stream>>>(
      yln16, wf1_16, NTOK, 1536, 384, b_fc1, h16, nullptr, nullptr);

  // 8) FC2 + residual -> d_out (f32)
  gemm_wmma_kernel<3><<<dim3(NTOK / 128, DIMC / 64), 256, 0, stream>>>(
      h16, wf2_16, NTOK, DIMC, 1536, b_fc2, nullptr, out, yres);
}
